// SelfAttention3d_10127532884647
// MI455X (gfx1250) — matrix operand order repaired and checked by reading, not compiled, not run on hardware
//
#include <hip/hip_runtime.h>
#include <stdint.h>

// ---------------------------------------------------------------------------
// Types for CDNA5 WMMA (wave32): 16x16x32 bf16 -> f32
// ---------------------------------------------------------------------------
typedef __attribute__((ext_vector_type(16))) __bf16 v16bf;
typedef __attribute__((ext_vector_type(8)))  float  v8f;

#define LDS_PITCH 40   // 32 + 8 pad; rows 80B -> 16B aligned, bank-conflict free

__device__ __forceinline__ __bf16 f2bf(float f) {
  union { float f; uint32_t u; } c; c.f = f;
  uint32_t u = c.u;
  uint32_t r = (u + 0x7FFFu + ((u >> 16) & 1u)) >> 16;   // RNE
  union { unsigned short s; __bf16 b; } o; o.s = (unsigned short)r;
  return o.b;
}
__device__ __forceinline__ float bf2f(__bf16 b) {
  union { __bf16 b; unsigned short s; } i; i.b = b;
  union { uint32_t u; float f; } o; o.u = ((uint32_t)i.s) << 16;
  return o.f;
}

union U128 { uint4 u; __bf16 b[8]; };

// ---------------------------------------------------------------------------
// fp32 -> bf16 cast
// ---------------------------------------------------------------------------
__global__ void cast_f32_bf16_kernel(const float* __restrict__ in,
                                     __bf16* __restrict__ out, long long n) {
  long long i = (long long)blockIdx.x * blockDim.x + threadIdx.x;
  if (i < n) out[i] = f2bf(in[i]);
}

// ---------------------------------------------------------------------------
// 2x2x2 max pool over [T=8,H=28,W=28] of k/v rows of the qkv buffer.
// ---------------------------------------------------------------------------
__global__ void pool222_kernel(const __bf16* __restrict__ qkv,
                               __bf16* __restrict__ kp,
                               __bf16* __restrict__ vp) {
  const int Lq = 6272, Lp = 784;
  int idx = blockIdx.x * blockDim.x + threadIdx.x;
  const int total = 4 * 320 * Lp;
  if (idx >= total) return;
  int lp = idx % Lp;
  int c  = (idx / Lp) % 320;
  int n  = idx / (Lp * 320);
  int wp = lp % 14, hp = (lp / 14) % 14, tp = lp / 196;
  const __bf16* src = qkv + (long long)n * 384 * Lq + (long long)(64 + c) * Lq;
  float m = -3.4e38f;
  #pragma unroll
  for (int dt = 0; dt < 2; ++dt)
    #pragma unroll
    for (int dh = 0; dh < 2; ++dh)
      #pragma unroll
      for (int dw = 0; dw < 2; ++dw) {
        int l = (2 * tp + dt) * 784 + (2 * hp + dh) * 28 + (2 * wp + dw);
        m = fmaxf(m, bf2f(src[l]));
      }
  if (c < 64)
    kp[(long long)n * 64 * Lp + (long long)c * Lp + lp] = f2bf(m);
  else
    vp[(long long)n * 256 * Lp + (long long)(c - 64) * Lp + lp] = f2bf(m);
}

// ---------------------------------------------------------------------------
// Row softmax: scores f32 [rows, 784] -> beta bf16 [rows, 784]
// ---------------------------------------------------------------------------
__global__ __launch_bounds__(256)
void softmax_kernel(const float* __restrict__ scores, __bf16* __restrict__ beta) {
  const int Lk = 784;
  long long row = blockIdx.x;
  const float* s = scores + row * Lk;
  __bf16* b = beta + row * Lk;
  __shared__ float red[256];
  int tid = threadIdx.x;

  float m = -3.4e38f;
  for (int i = tid; i < Lk; i += 256) m = fmaxf(m, s[i]);
  red[tid] = m; __syncthreads();
  for (int off = 128; off > 0; off >>= 1) {
    if (tid < off) red[tid] = fmaxf(red[tid], red[tid + off]);
    __syncthreads();
  }
  m = red[0]; __syncthreads();

  float sum = 0.f;
  for (int i = tid; i < Lk; i += 256) sum += __expf(s[i] - m);
  red[tid] = sum; __syncthreads();
  for (int off = 128; off > 0; off >>= 1) {
    if (tid < off) red[tid] += red[tid + off];
    __syncthreads();
  }
  float inv = 1.0f / red[0];
  for (int i = tid; i < Lk; i += 256) b[i] = f2bf(__expf(s[i] - m) * inv);
}

// ---------------------------------------------------------------------------
// Tiled WMMA GEMM (bf16 in, f32 accumulate), compile-time specialized:
//   AK : A element (m,k) contiguous in k  -> async global->LDS staging
//   BK : B^T element (n,k) contiguous in k -> async global->LDS staging
//   EPI: 0 = store bf16, 1 = store f32, 2 = out = y*acc + Xres
// Macro tile 128x128, K step 32, 256 threads = 8 waves (4x2 of 32x64).
// Double-buffered LDS: async loads for tile t+1 overlap WMMAs on tile t;
// one s_wait_asynccnt + one barrier per K-step.
// ---------------------------------------------------------------------------
template<bool AK, bool BK, int EPI>
__global__ __launch_bounds__(256)
void gemm_wmma_bf16_kernel(const __bf16* __restrict__ A, long long ldaR, long long ldaC, long long strideA,
                           const __bf16* __restrict__ B, long long ldbR, long long ldbC, long long strideB,
                           void* __restrict__ Cout, long long ldc, long long strideC,
                           int M, int N, int K,
                           const float* __restrict__ Xres, const float* __restrict__ yptr) {
  __shared__ __align__(16) __bf16 sA[2][128 * LDS_PITCH];   // [m][k]
  __shared__ __align__(16) __bf16 sB[2][128 * LDS_PITCH];   // [n][k] (B transposed)

  const int tid  = threadIdx.x;
  const int wave = tid >> 5;
  const int lane = tid & 31;
  const int lrow = lane & 15;
  const int half = lane >> 4;

  const int bm = blockIdx.y * 128;
  const int bn = blockIdx.x * 128;
  const int batch = blockIdx.z;

  const __bf16* Ab = A + (long long)batch * strideA;
  const __bf16* Bb = B + (long long)batch * strideB;

  const int wm = (wave & 3) * 32;
  const int wn = (wave >> 2) * 64;

  // ---- tile staging (issues async loads and/or LDS stores; no waits) ----
  auto stage = [&](int k0, __bf16* sAb, __bf16* sBb) {
    // ---------------- A tile (128 x 32) ----------------
    if constexpr (AK) {
      #pragma unroll
      for (int j = 0; j < 2; ++j) {
        int c   = tid + j * 256;
        int mm  = c >> 2;
        int kk8 = (c & 3) << 3;
        int gm  = bm + mm, gk = k0 + kk8;
        __bf16* dst = sAb + mm * LDS_PITCH + kk8;
        if (gm < M && gk + 8 <= K) {
          const __bf16* gp = Ab + (long long)gm * ldaR + gk;
          uint32_t ldsOff = (uint32_t)(uintptr_t)dst;
          uint64_t ga = (uint64_t)(uintptr_t)gp;
          asm volatile("global_load_async_to_lds_b128 %0, %1, off"
                       :: "v"(ldsOff), "v"(ga) : "memory");
        } else {
          *(uint4*)dst = make_uint4(0u, 0u, 0u, 0u);
        }
      }
    } else {
      // m-contiguous (ldaR==1): b128 load along m, b16 scatter to LDS
      #pragma unroll
      for (int j = 0; j < 2; ++j) {
        int c   = tid + j * 256;
        int kk  = c >> 4;
        int mm8 = (c & 15) << 3;
        int gm  = bm + mm8, gk = k0 + kk;
        U128 d; d.u = make_uint4(0u, 0u, 0u, 0u);
        if (gm + 8 <= M && gk < K)
          d.u = *(const uint4*)(Ab + (long long)gm * ldaR + (long long)gk * ldaC);
        #pragma unroll
        for (int i = 0; i < 8; ++i)
          sAb[(mm8 + i) * LDS_PITCH + kk] = d.b[i];
      }
    }
    // ---------------- B tile (32 x 128 -> [n][k]) ----------------
    if constexpr (BK) {
      #pragma unroll
      for (int j = 0; j < 2; ++j) {
        int c   = tid + j * 256;
        int nn  = c >> 2;
        int kk8 = (c & 3) << 3;
        int gn  = bn + nn, gk = k0 + kk8;
        __bf16* dst = sBb + nn * LDS_PITCH + kk8;
        if (gn < N && gk + 8 <= K) {
          const __bf16* gp = Bb + (long long)gn * ldbC + gk;
          uint32_t ldsOff = (uint32_t)(uintptr_t)dst;
          uint64_t ga = (uint64_t)(uintptr_t)gp;
          asm volatile("global_load_async_to_lds_b128 %0, %1, off"
                       :: "v"(ldsOff), "v"(ga) : "memory");
        } else {
          *(uint4*)dst = make_uint4(0u, 0u, 0u, 0u);
        }
      }
    } else {
      // n-contiguous (ldbC==1): b128 load along n, transpose via b16 scatter
      #pragma unroll
      for (int j = 0; j < 2; ++j) {
        int c   = tid + j * 256;
        int kk  = c >> 4;
        int nn8 = (c & 15) << 3;
        int gn  = bn + nn8, gk = k0 + kk;
        U128 d; d.u = make_uint4(0u, 0u, 0u, 0u);
        if (gk < K && gn + 8 <= N)
          d.u = *(const uint4*)(Bb + (long long)gk * ldbR + gn);
        #pragma unroll
        for (int i = 0; i < 8; ++i)
          sBb[(nn8 + i) * LDS_PITCH + kk] = d.b[i];
      }
    }
  };

  v8f acc[2][4];
  #pragma unroll
  for (int i = 0; i < 2; ++i)
    #pragma unroll
    for (int j = 0; j < 4; ++j)
      #pragma unroll
      for (int r = 0; r < 8; ++r) acc[i][j][r] = 0.0f;

  const int KT = (K + 31) >> 5;

  // ---- prologue: stage tile 0 ----
  stage(0, sA[0], sB[0]);
  if constexpr (AK || BK)
    asm volatile("s_wait_asynccnt 0x0" ::: "memory");
  __syncthreads();

  for (int kt = 0; kt < KT; ++kt) {
    const int cur = kt & 1;
    const __bf16* sAc = sA[cur];
    const __bf16* sBc = sB[cur];

    // stage next tile into the other buffer (overlaps with WMMA below)
    if (kt + 1 < KT)
      stage((kt + 1) << 5, sA[cur ^ 1], sB[cur ^ 1]);

    // ---- load fragments per ISA 7.12.2 layouts ----
    v16bf afrag[2], bfrag[4];
    #pragma unroll
    for (int fm = 0; fm < 2; ++fm) {
      int m = wm + fm * 16 + lrow;
      #pragma unroll
      for (int i = 0; i < 8; ++i) {
        // A (16-bit, 16x32): VGPR i -> K = (i<4?0:16) + 8*half + 2*(i&3)
        int k = ((i & 4) << 2) + half * 8 + (i & 3) * 2;
        union { uint32_t u; __bf16 b[2]; } cv;
        cv.u = *(const uint32_t*)(sAc + m * LDS_PITCH + k);
        afrag[fm][2 * i]     = cv.b[0];
        afrag[fm][2 * i + 1] = cv.b[1];
      }
    }
    #pragma unroll
    for (int fn = 0; fn < 4; ++fn) {
      int nn = wn + fn * 16 + lrow;
      #pragma unroll
      for (int i = 0; i < 8; ++i) {
        // B (32x16): VGPR i -> K = (i<4?0:16) + 8*half + 2*(i&3)
        int k = ((i & 4) << 2) + half * 8 + (i & 3) * 2;
        union { uint32_t u; __bf16 b[2]; } cv;
        cv.u = *(const uint32_t*)(sBc + nn * LDS_PITCH + k);
        bfrag[fn][2 * i]     = cv.b[0];
        bfrag[fn][2 * i + 1] = cv.b[1];
      }
    }

    // ---- 8 WMMAs per wave per K step ----
    #pragma unroll
    for (int fm = 0; fm < 2; ++fm)
      #pragma unroll
      for (int fn = 0; fn < 4; ++fn)
        acc[fm][fn] = __builtin_amdgcn_wmma_f32_16x16x32_bf16(
            false, afrag[fm], false, bfrag[fn],
            (short)0, acc[fm][fn], false, false);

    // my async writes into the next buffer must land before the barrier;
    // they had the whole compute phase to complete.
    if constexpr (AK || BK)
      asm volatile("s_wait_asynccnt 0x0" ::: "memory");
    __syncthreads();
  }

  // ---- epilogue: lane = N%16, VGPR r -> M = r + 8*half ----
  const long long cb = (long long)batch * strideC;
  float yv = 0.0f;
  if constexpr (EPI == 2) yv = *yptr;
  #pragma unroll
  for (int fm = 0; fm < 2; ++fm) {
    #pragma unroll
    for (int fn = 0; fn < 4; ++fn) {
      int gn = bn + wn + fn * 16 + lrow;
      if (gn >= N) continue;
      #pragma unroll
      for (int r = 0; r < 8; ++r) {
        int gm = bm + wm + fm * 16 + r + half * 8;
        if (gm >= M) continue;
        float v = acc[fm][fn][r];
        long long off = cb + (long long)gm * ldc + gn;
        if constexpr (EPI == 0)      ((__bf16*)Cout)[off] = f2bf(v);
        else if constexpr (EPI == 1) ((float*)Cout)[off]  = v;
        else                         ((float*)Cout)[off]  = yv * v + Xres[off];
      }
    }
  }
}

// ---------------------------------------------------------------------------
// Host launcher
// ---------------------------------------------------------------------------
extern "C" void kernel_launch(void* const* d_in, const int* in_sizes, int n_in,
                              void* d_out, int out_size, void* d_ws, size_t ws_size,
                              hipStream_t stream) {
  (void)in_sizes; (void)n_in; (void)out_size; (void)ws_size;
  const int NB = 4, C = 512, T = 8, H = 28, W = 28;
  const int CQ = 64, CV = 256;
  const int L  = T * H * W;          // 6272
  const int Lp = (T/2)*(H/2)*(W/2);  // 784

  const float* x  = (const float*)d_in[0];
  const float* Wq = (const float*)d_in[1];
  const float* Wk = (const float*)d_in[2];
  const float* Wv = (const float*)d_in[3];
  const float* Wz = (const float*)d_in[4];
  const float* y  = (const float*)d_in[5];
  float* out = (float*)d_out;

  char* ws = (char*)d_ws;
  auto al = [](size_t s) { return (s + 255) & ~(size_t)255; };
  size_t off = 0;
  __bf16* xb    = (__bf16*)(ws + off); off += al((size_t)NB * C * L * 2);
  __bf16* wqkv  = (__bf16*)(ws + off); off += al((size_t)384 * 512 * 2);
  __bf16* wzb   = (__bf16*)(ws + off); off += al((size_t)512 * 256 * 2);
  __bf16* qkv   = (__bf16*)(ws + off); off += al((size_t)NB * 384 * L * 2);
  __bf16* kp    = (__bf16*)(ws + off); off += al((size_t)NB * CQ * Lp * 2);
  __bf16* vp    = (__bf16*)(ws + off); off += al((size_t)NB * CV * Lp * 2);
  float*  sc    = (float*)(ws + off);  off += al((size_t)NB * L * Lp * 4);
  __bf16* beta  = (__bf16*)(ws + off); off += al((size_t)NB * L * Lp * 2);
  __bf16* att   = (__bf16*)(ws + off); off += al((size_t)NB * CV * L * 2);

  // --- casts to bf16 ---
  long long nx = (long long)NB * C * L;
  cast_f32_bf16_kernel<<<(unsigned)((nx + 255) / 256), 256, 0, stream>>>(x, xb, nx);
  cast_f32_bf16_kernel<<<(64 * 512 + 255) / 256, 256, 0, stream>>>(Wq, wqkv, 64 * 512);
  cast_f32_bf16_kernel<<<(64 * 512 + 255) / 256, 256, 0, stream>>>(Wk, wqkv + 64 * 512, 64 * 512);
  cast_f32_bf16_kernel<<<(256 * 512 + 255) / 256, 256, 0, stream>>>(Wv, wqkv + 128 * 512, 256 * 512);
  cast_f32_bf16_kernel<<<(512 * 256 + 255) / 256, 256, 0, stream>>>(Wz, wzb, 512 * 256);

  dim3 blk(256);

  // --- G1: QKV projection. [384,512] x [512,6272] per batch -> bf16 qkv ---
  gemm_wmma_bf16_kernel<true, false, 0><<<dim3(49, 3, 4), blk, 0, stream>>>(
      wqkv, 512, 1, 0,
      xb, L, 1, (long long)C * L,
      qkv, L, (long long)384 * L,
      384, L, 512, nullptr, nullptr);

  // --- maxpool 2x2x2 on k and v channels ---
  int ptot = NB * (CQ + CV) * Lp;
  pool222_kernel<<<(ptot + 255) / 256, 256, 0, stream>>>(qkv, kp, vp);

  // --- G2: scores = q^T k. [6272,64] x [64,784] per batch -> f32 ---
  gemm_wmma_bf16_kernel<false, false, 1><<<dim3(7, 49, 4), blk, 0, stream>>>(
      qkv, 1, L, (long long)384 * L,          // A[m,k] = q[k*L + m]
      kp, Lp, 1, (long long)CQ * Lp,
      sc, Lp, (long long)L * Lp,
      L, Lp, CQ, nullptr, nullptr);

  // --- softmax over Lk rows -> beta bf16 ---
  softmax_kernel<<<NB * L, 256, 0, stream>>>(sc, beta);

  // --- G3: att = v * beta^T. [256,784] x [784,6272] per batch -> bf16 ---
  gemm_wmma_bf16_kernel<true, true, 0><<<dim3(49, 2, 4), blk, 0, stream>>>(
      vp, Lp, 1, (long long)CV * Lp,
      beta, 1, Lp, (long long)L * Lp,         // B[k,n] = beta[n*Lp + k]
      att, L, (long long)CV * L,
      CV, L, Lp, nullptr, nullptr);

  // --- G4: out = y * (Wz att) + x. [512,256] x [256,6272] per batch -> f32 ---
  gemm_wmma_bf16_kernel<true, false, 2><<<dim3(49, 4, 4), blk, 0, stream>>>(
      wzb, 256, 1, 0,
      att, L, 1, (long long)CV * L,
      out, L, (long long)C * L,
      C, L, CV, x, y);
}
